// MPNNetDrop_44220983279846
// MI455X (gfx1250) — matrix operand order repaired and checked by reading, not compiled, not run on hardware
//
#include <hip/hip_runtime.h>
#include <hip/hip_bf16.h>

#define N_NODES 30000
#define N_EDGES 60000
#define N_GRAPH 1200
#define DIM 64
#define FEAT 14
#define NPG 25

typedef __attribute__((ext_vector_type(16))) _Float16 v16h;
typedef __attribute__((ext_vector_type(8)))  float    v8f;

union V16 { v16h v; uint4 u[2]; };

// ---- CDNA5 async copy: global -> LDS (ASYNCcnt-tracked), GV addressing mode ----
__device__ __forceinline__ void async_ld128(unsigned ldsoff, const void* gaddr) {
  asm volatile("global_load_async_to_lds_b128 %0, %1, off"
               :: "v"(ldsoff), "v"((unsigned long long)(uintptr_t)gaddr)
               : "memory");
}
__device__ __forceinline__ void wait_async0() {
  asm volatile("s_wait_asynccnt 0x0" ::: "memory");
}

// ---------------- utility kernels ----------------

__global__ void zero_f32(float* __restrict__ p, int n) {
  int i = blockIdx.x * blockDim.x + threadIdx.x;
  if (i < n) p[i] = 0.0f;
}

__global__ void cvt_f16_kernel(const float* __restrict__ s, _Float16* __restrict__ d, int n) {
  int i = blockIdx.x * blockDim.x + threadIdx.x;
  if (i < n) d[i] = (_Float16)s[i];
}

__global__ void deg_kernel(const int* __restrict__ tgt, float* __restrict__ deg) {
  int e = blockIdx.x * blockDim.x + threadIdx.x;
  if (e < N_EDGES) atomicAdd(&deg[tgt[e]], 1.0f);
}

// out = relu(x @ W_lin0^T + b), x:[N,14], W:[64,14]
__global__ void lin0_kernel(const float* __restrict__ x, const float* __restrict__ W,
                            const float* __restrict__ b, float* __restrict__ out) {
  int i = blockIdx.x * blockDim.x + threadIdx.x;
  if (i >= N_NODES * DIM) return;
  int n = i >> 6, d = i & 63;
  float acc = b[d];
  const float* xr = x + (size_t)n * FEAT;
  const float* wr = W + (size_t)d * FEAT;
  #pragma unroll
  for (int k = 0; k < FEAT; ++k) acc += xr[k] * wr[k];
  out[i] = fmaxf(acc, 0.0f);
}

// h1 = relu(edge_attr @ W_h1^T + b_h1) -> f16, ea:[E,4], W:[128,4]
__global__ void h1_kernel(const float* __restrict__ ea, const float* __restrict__ W,
                          const float* __restrict__ b, _Float16* __restrict__ h1) {
  int i = blockIdx.x * blockDim.x + threadIdx.x;
  if (i >= N_EDGES * 128) return;
  int e = i >> 7, c = i & 127;
  float acc = b[c];
  const float* er = ea + (size_t)e * 4;
  const float* wr = W + (size_t)c * 4;
  #pragma unroll
  for (int k = 0; k < 4; ++k) acc += er[k] * wr[k];
  h1[i] = (_Float16)fmaxf(acc, 0.0f);
}

// ---------------- fused edge-network GEMM + message contraction (WMMA) ----------------
// Per 16-edge tile: EW[e, i*64+o] = sum_k h1[e,k] * W_h2[i*64+o, k] + b_h2[i*64+o]
// msg[e,o] = sum_i out[src[e],i] * EW[e, i*64+o]; aggr[tgt[e],o] += msg[e,o]/deg
__global__ __launch_bounds__(256) void msg_kernel(
    const _Float16* __restrict__ h1, const _Float16* __restrict__ wh2,
    const float* __restrict__ bh2, const float* __restrict__ out,
    const int* __restrict__ src, const int* __restrict__ tgt,
    const float* __restrict__ deg, float* __restrict__ aggr) {
  __shared__ __align__(16) _Float16 ldsB[64 * 128];   // 16 KB: W_h2 rows for current i
  __shared__ __align__(16) float    ldsO[128 * DIM];  // 32 KB: gathered out[src] for 128 edges

  const int tid  = threadIdx.x;
  const int lane = tid & 31;
  const int w    = tid >> 5;               // wave id 0..7
  const int blk0 = blockIdx.x * 128;       // first edge of workgroup
  const int e0   = blk0 + w * 16;          // first edge of this wave
  const bool active = (e0 < N_EDGES);      // E % 16 == 0, so waves are all-or-nothing

  const unsigned ldsO_base = (unsigned)(uintptr_t)ldsO;
  const unsigned ldsB_base = (unsigned)(uintptr_t)ldsB;

  // async gather: out[src[e]] rows (64 f32 each) for the block's 128 edges -> LDS
  #pragma unroll
  for (int r = 0; r < 8; ++r) {
    int idx = r * 256 + tid;               // 2048 x 16B total
    int el = idx >> 4, q = idx & 15;
    int e = blk0 + el;
    if (e < N_EDGES)
      async_ld128(ldsO_base + (unsigned)idx * 16u,
                  (const char*)(out + (size_t)src[e] * DIM) + (size_t)q * 16);
  }

  // A operand (WMMA 16x32 f16 layout): lanes0-15 row m, K 0..7 / 16..23; lanes16-31 K 8..15 / 24..31
  const int m    = lane & 15;
  const int hi8  = (lane < 16) ? 0 : 8;
  const int hiB  = hi8 * 2;                // 0 or 16, unused: B uses the hi8 split of A
  const int col  = lane & 15;
  V16 A[4];
  if (active) {
    const _Float16* hr = h1 + (size_t)(e0 + m) * 128;
    #pragma unroll
    for (int kc = 0; kc < 4; ++kc) {
      A[kc].u[0] = *(const uint4*)(hr + kc * 32 + hi8);
      A[kc].u[1] = *(const uint4*)(hr + kc * 32 + 16 + hi8);
    }
  }

  float msg[4][8];
  #pragma unroll
  for (int c = 0; c < 4; ++c)
    #pragma unroll
    for (int r = 0; r < 8; ++r) msg[c][r] = 0.0f;

  for (int i = 0; i < DIM; ++i) {
    __syncthreads();   // everyone done reading previous ldsB slab
    // async stage of W_h2 f16 rows [i*64, i*64+64) -> LDS (shared by all 8 waves)
    const char* wsrc = (const char*)(wh2 + (size_t)i * 64 * 128);
    #pragma unroll
    for (int r = 0; r < 4; ++r) {
      int idx = r * 256 + tid;             // 1024 x 16B = 16 KB
      async_ld128(ldsB_base + (unsigned)idx * 16u, wsrc + (size_t)idx * 16);
    }
    wait_async0();                          // this wave's async copies landed (incl. ldsO on i==0)
    __syncthreads();                        // all waves' copies landed
    if (!active) continue;

    float s[8];
    #pragma unroll
    for (int r = 0; r < 8; ++r)
      s[r] = ldsO[(size_t)(w * 16 + r + hi8) * DIM + i];

    #pragma unroll
    for (int c = 0; c < 4; ++c) {          // o-subtile: cols i*64 + c*16 + [0,16)
      v8f acc = {};
      #pragma unroll
      for (int kc = 0; kc < 4; ++kc) {
        V16 Bv;
        const _Float16* bp = ldsB + (size_t)(c * 16 + col) * 128 + kc * 32 + hi8;
        Bv.u[0] = *(const uint4*)bp;
        Bv.u[1] = *(const uint4*)(bp + 16);
        acc = __builtin_amdgcn_wmma_f32_16x16x32_f16(
            false, A[kc].v, false, Bv.v, (short)0, acc, false, false);
      }
      float bias = bh2[i * 64 + c * 16 + col];
      #pragma unroll
      for (int r = 0; r < 8; ++r)
        msg[c][r] += s[r] * (acc[r] + bias);
    }
  }

  if (active) {
    #pragma unroll
    for (int r = 0; r < 8; ++r) {
      int e = e0 + r + hi8;
      int t = tgt[e];
      float invd = 1.0f / fmaxf(deg[t], 1.0f);
      #pragma unroll
      for (int c = 0; c < 4; ++c)
        atomicAdd(&aggr[(size_t)t * DIM + c * 16 + col], msg[c][r] * invd);
    }
  }
}

// ---------------- node update ----------------

__device__ __forceinline__ float dot64(const float* __restrict__ a, const float* __restrict__ w) {
  float acc = 0.0f;
  #pragma unroll
  for (int k = 0; k < 64; ++k) acc += a[k] * w[k];
  return acc;
}

// m = relu(out @ W_root + aggr + b_conv), W_root row-major [64,64]
__global__ void convm_kernel(const float* __restrict__ out, const float* __restrict__ aggr,
                             const float* __restrict__ Wroot, const float* __restrict__ bconv,
                             float* __restrict__ mbuf) {
  int i = blockIdx.x * blockDim.x + threadIdx.x;
  if (i >= N_NODES * DIM) return;
  int n = i >> 6, d = i & 63;
  const float* orow = out + (size_t)n * DIM;
  float acc = aggr[i] + bconv[d];
  #pragma unroll
  for (int j = 0; j < DIM; ++j) acc += orow[j] * Wroot[j * DIM + d];
  mbuf[i] = fmaxf(acc, 0.0f);
}

// GRU cell (gate order r,z,n), h_new written to separate buffer
__global__ void gru_kernel(const float* __restrict__ m, const float* __restrict__ hold,
                           const float* __restrict__ Wih, const float* __restrict__ Whh,
                           const float* __restrict__ bih, const float* __restrict__ bhh,
                           float* __restrict__ hnew) {
  int i = blockIdx.x * blockDim.x + threadIdx.x;
  if (i >= N_NODES * DIM) return;
  int n = i >> 6, d = i & 63;
  const float* mr = m + (size_t)n * DIM;
  const float* hr = hold + (size_t)n * DIM;
  float ir  = bih[d]       + dot64(mr, Wih + (size_t)d * DIM);
  float iz  = bih[64 + d]  + dot64(mr, Wih + (size_t)(64 + d) * DIM);
  float inn = bih[128 + d] + dot64(mr, Wih + (size_t)(128 + d) * DIM);
  float hrr = bhh[d]       + dot64(hr, Whh + (size_t)d * DIM);
  float hz  = bhh[64 + d]  + dot64(hr, Whh + (size_t)(64 + d) * DIM);
  float hn  = bhh[128 + d] + dot64(hr, Whh + (size_t)(128 + d) * DIM);
  float r = 1.0f / (1.0f + expf(-(ir + hrr)));
  float z = 1.0f / (1.0f + expf(-(iz + hz)));
  float nn = tanhf(inn + r * hn);
  hnew[i] = (1.0f - z) * nn + z * hr[d];
}

// ---------------- Set2Set ----------------

__global__ void lstm_kernel(const float* __restrict__ qstar, const float* __restrict__ hl,
                            const float* __restrict__ cl, const float* __restrict__ Wlih,
                            const float* __restrict__ Wlhh, const float* __restrict__ blih,
                            const float* __restrict__ blhh,
                            float* __restrict__ hln, float* __restrict__ cln) {
  int i = blockIdx.x * blockDim.x + threadIdx.x;
  if (i >= N_GRAPH * DIM) return;
  int b = i >> 6, d = i & 63;
  const float* q = qstar + (size_t)b * 128;
  const float* h = hl + (size_t)b * DIM;
  float g[4];
  #pragma unroll
  for (int gg = 0; gg < 4; ++gg) {
    int row = gg * 64 + d;
    float acc = blih[row] + blhh[row];
    const float* wi = Wlih + (size_t)row * 128;
    #pragma unroll 8
    for (int k = 0; k < 128; ++k) acc += q[k] * wi[k];
    acc += dot64(h, Wlhh + (size_t)row * DIM);
    g[gg] = acc;
  }
  float ig = 1.0f / (1.0f + expf(-g[0]));
  float fg = 1.0f / (1.0f + expf(-g[1]));
  float gv = tanhf(g[2]);
  float og = 1.0f / (1.0f + expf(-g[3]));
  float c = fg * cl[i] + ig * gv;
  cln[i] = c;
  hln[i] = og * tanhf(c);
}

// attention + softmax + weighted readout per graph (25 contiguous nodes per graph)
__global__ void attend_kernel(const float* __restrict__ out, const float* __restrict__ hl,
                              float* __restrict__ qstar) {
  int g = blockIdx.x;
  int t = threadIdx.x;  // 64 threads
  __shared__ float sh_e[NPG];
  __shared__ float sh_q[DIM];
  __shared__ float sh_sum;
  sh_q[t] = hl[(size_t)g * DIM + t];
  __syncthreads();
  if (t < NPG) {
    const float* orow = out + (size_t)(g * NPG + t) * DIM;
    float acc = 0.0f;
    #pragma unroll
    for (int d = 0; d < DIM; ++d) acc += orow[d] * sh_q[d];
    sh_e[t] = acc;
  }
  __syncthreads();
  if (t == 0) {
    float mx = sh_e[0];
    for (int j = 1; j < NPG; ++j) mx = fmaxf(mx, sh_e[j]);
    float s = 0.0f;
    for (int j = 0; j < NPG; ++j) { float a = expf(sh_e[j] - mx); sh_e[j] = a; s += a; }
    sh_sum = s;
  }
  __syncthreads();
  float r = 0.0f;
  #pragma unroll
  for (int j = 0; j < NPG; ++j) r += sh_e[j] * out[(size_t)(g * NPG + j) * DIM + t];
  qstar[(size_t)g * 128 + t] = sh_q[t];
  qstar[(size_t)g * 128 + DIM + t] = r / sh_sum;
}

__global__ void readout_kernel(const float* __restrict__ qstar, const float* __restrict__ Wlin1,
                               const float* __restrict__ blin1, const float* __restrict__ Wlin2,
                               const float* __restrict__ blin2, float* __restrict__ outp) {
  int g = blockIdx.x, d = threadIdx.x;  // 64 threads
  __shared__ float sh[DIM];
  const float* q = qstar + (size_t)g * 128;
  const float* w = Wlin1 + (size_t)d * 128;
  float acc = blin1[d];
  #pragma unroll 8
  for (int k = 0; k < 128; ++k) acc += q[k] * w[k];
  acc = fmaxf(acc, 0.0f);
  sh[d] = acc * Wlin2[d];
  __syncthreads();
  if (d == 0) {
    float s = blin2[0];
    for (int j = 0; j < DIM; ++j) s += sh[j];
    outp[g] = s;
  }
}

// ---------------- launcher ----------------

extern "C" void kernel_launch(void* const* d_in, const int* in_sizes, int n_in,
                              void* d_out, int out_size, void* d_ws, size_t ws_size,
                              hipStream_t stream) {
  (void)in_sizes; (void)n_in; (void)out_size; (void)ws_size;
  const float* x     = (const float*)d_in[0];
  const float* ea    = (const float*)d_in[1];
  const float* Wlin0 = (const float*)d_in[2];
  const float* blin0 = (const float*)d_in[3];
  const float* Wh1   = (const float*)d_in[4];
  const float* bh1   = (const float*)d_in[5];
  const float* Wh2   = (const float*)d_in[6];
  const float* bh2   = (const float*)d_in[7];
  const float* Wroot = (const float*)d_in[8];
  const float* bconv = (const float*)d_in[9];
  const float* Wih   = (const float*)d_in[10];
  const float* Whh   = (const float*)d_in[11];
  const float* bih   = (const float*)d_in[12];
  const float* bhh   = (const float*)d_in[13];
  const float* Wlih  = (const float*)d_in[14];
  const float* Wlhh  = (const float*)d_in[15];
  const float* blih  = (const float*)d_in[16];
  const float* blhh  = (const float*)d_in[17];
  const float* Wlin1 = (const float*)d_in[18];
  const float* blin1 = (const float*)d_in[19];
  const float* Wlin2 = (const float*)d_in[20];
  const float* blin2 = (const float*)d_in[21];
  const int*   ei    = (const int*)d_in[22];
  const int* srcp = ei;
  const int* tgtp = ei + N_EDGES;
  float* outp = (float*)d_out;

  char* ws = (char*)d_ws;
  size_t off = 0;
  auto alloc = [&](size_t bytes) -> void* {
    void* p = ws + off;
    off += (bytes + 255) & ~(size_t)255;
    return p;
  };
  float*    outA  = (float*)alloc((size_t)N_NODES * DIM * 4);
  float*    outB  = (float*)alloc((size_t)N_NODES * DIM * 4);
  float*    aggr  = (float*)alloc((size_t)N_NODES * DIM * 4);
  float*    mbuf  = (float*)alloc((size_t)N_NODES * DIM * 4);
  _Float16* h1f   = (_Float16*)alloc((size_t)N_EDGES * 128 * 2);
  _Float16* wh2f  = (_Float16*)alloc((size_t)4096 * 128 * 2);
  float*    degb  = (float*)alloc((size_t)N_NODES * 4);
  float*    qstar = (float*)alloc((size_t)N_GRAPH * 128 * 4);
  float*    hlA   = (float*)alloc((size_t)N_GRAPH * DIM * 4);
  float*    hlB   = (float*)alloc((size_t)N_GRAPH * DIM * 4);
  float*    clA   = (float*)alloc((size_t)N_GRAPH * DIM * 4);
  float*    clB   = (float*)alloc((size_t)N_GRAPH * DIM * 4);

  const int ND = N_NODES * DIM;                 // 1,920,000
  const int gND = (ND + 255) / 256;             // 7500

  // phase A: precompute
  lin0_kernel<<<gND, 256, 0, stream>>>(x, Wlin0, blin0, outA);
  h1_kernel<<<(N_EDGES * 128 + 255) / 256, 256, 0, stream>>>(ea, Wh1, bh1, h1f);
  cvt_f16_kernel<<<(4096 * 128 + 255) / 256, 256, 0, stream>>>(Wh2, wh2f, 4096 * 128);
  zero_f32<<<(N_NODES + 255) / 256, 256, 0, stream>>>(degb, N_NODES);
  deg_kernel<<<(N_EDGES + 255) / 256, 256, 0, stream>>>(tgtp, degb);

  // phase B: 3 message-passing + GRU iterations
  float* cur = outA;
  float* nxt = outB;
  const int msgGrid = (N_EDGES / 16 + 7) / 8;   // 469 workgroups, 8 waves each
  for (int it = 0; it < 3; ++it) {
    zero_f32<<<gND, 256, 0, stream>>>(aggr, ND);
    msg_kernel<<<msgGrid, 256, 0, stream>>>(h1f, wh2f, bh2, cur, srcp, tgtp, degb, aggr);
    convm_kernel<<<gND, 256, 0, stream>>>(cur, aggr, Wroot, bconv, mbuf);
    gru_kernel<<<gND, 256, 0, stream>>>(mbuf, cur, Wih, Whh, bih, bhh, nxt);
    float* tmp = cur; cur = nxt; nxt = tmp;
  }

  // phase C: Set2Set (3 steps)
  zero_f32<<<(N_GRAPH * 128 + 255) / 256, 256, 0, stream>>>(qstar, N_GRAPH * 128);
  zero_f32<<<(N_GRAPH * DIM + 255) / 256, 256, 0, stream>>>(hlA, N_GRAPH * DIM);
  zero_f32<<<(N_GRAPH * DIM + 255) / 256, 256, 0, stream>>>(clA, N_GRAPH * DIM);
  float *hc = hlA, *cc = clA, *hn = hlB, *cn = clB;
  for (int s = 0; s < 3; ++s) {
    lstm_kernel<<<(N_GRAPH * DIM + 255) / 256, 256, 0, stream>>>(
        qstar, hc, cc, Wlih, Wlhh, blih, blhh, hn, cn);
    attend_kernel<<<N_GRAPH, 64, 0, stream>>>(cur, hn, qstar);
    float* t1 = hc; hc = hn; hn = t1;
    float* t2 = cc; cc = cn; cn = t2;
  }

  // readout
  readout_kernel<<<N_GRAPH, 64, 0, stream>>>(qstar, Wlin1, blin1, Wlin2, blin2, outp);
}
